// Conv_46179488366843
// MI455X (gfx1250) — matrix operand order repaired and checked by reading, not compiled, not run on hardware
//
#include <hip/hip_runtime.h>
#include <hip/hip_bf16.h>
#include <math.h>

// ---------------------------------------------------------------------------
// Equivariant edge-message conv for MI455X (gfx1250), wave32 + WMMA bf16.
//   E=100000 edges, N=10000 nodes, M=16, ED=32, HID=128, WNUM=1792.
// Pipeline: pack weights -> fused edge MLP (WMMA) + contraction + atomics
//           -> per-node finalize (mean, residual, gate).
// ---------------------------------------------------------------------------

typedef __attribute__((ext_vector_type(16))) __bf16 bf16x16;
typedef __attribute__((ext_vector_type(8)))  __bf16 bf16x8;
typedef __attribute__((ext_vector_type(8)))  float  f32x8;
typedef __attribute__((ext_vector_type(4)))  float  f32x4;

#define ED   32
#define HID  128
#define WNUM 1792
#define NCHUNK 112          // WNUM / 16

// constants: A0=1/sqrt(2M), A1=1/sqrt(3M), INV3=1/sqrt3, INV2=1/sqrt2, M=16
#define C_A0   0.17677669529663687f
#define C_A1   0.14433756729740643f
#define C_INV3 0.57735026918962576f
#define C_INV2 0.70710678118654752f

// ------------------------------ zero workspace -----------------------------
__global__ void zero_f32(float* __restrict__ p, long long n) {
  long long i = (long long)blockIdx.x * blockDim.x + threadIdx.x;
  if (i < n) p[i] = 0.0f;
}

// --------------------------- weight pack (f32->bf16) -----------------------
// B-tile layout (32x16 K-major WMMA B operand): element (tile, lane, h)
//   col = tileN*16 + (lane&15);  K = (lane>>4)*8 + h + (h>=8 ? 8 : 0)   (h = 0..15)
__global__ void pack_weights(const float* __restrict__ We1,
                             const float* __restrict__ We2,
                             __bf16* __restrict__ We1pk,
                             __bf16* __restrict__ We2pk) {
  int t = blockIdx.x * blockDim.x + threadIdx.x;
  if (t < 8 * 32 * 16) {                       // We1: 8 N-tiles of (32x16)
    int h = t & 15, lane = (t >> 4) & 31, n1 = t >> 9;
    int colg = n1 * 16 + (lane & 15);
    int K = ((lane >> 4) << 3) + h + (h >= 8 ? 8 : 0);
    We1pk[t] = (__bf16)We1[K * HID + colg];
  }
  int t2 = t - 8 * 32 * 16;
  if (t2 >= 0 && t2 < NCHUNK * 4 * 32 * 16) {  // We2: 112 N-tiles x 4 K-steps
    int h = t2 & 15, lane = (t2 >> 4) & 31, tl = t2 >> 9;
    int n = tl >> 2, kk = tl & 3;
    int colg = n * 16 + (lane & 15);
    int K = kk * 32 + ((lane >> 4) << 3) + h + (h >= 8 ? 8 : 0);
    We2pk[t2] = (__bf16)We2[K * WNUM + colg];
  }
}

// ------------------------------- edge kernel -------------------------------
struct __align__(64) WaveLds {
  float coef0[2][16][16];     // [ss|vv0][m][edge]        (A0 baked in)
  float coef1[3][3][16][16];  // [sv|vs|vv1][c][m][edge]  (A1 baked in)
  __bf16 H[16][HID];          // relu(EA@We1+be1), row-major per edge
};

__launch_bounds__(128)
__global__ void edge_kernel(const int*   __restrict__ dst,
                            const float* __restrict__ x_src,
                            const float* __restrict__ sh,
                            const float* __restrict__ edge_attr,
                            const float* __restrict__ be1,
                            const float* __restrict__ be2,
                            const __bf16* __restrict__ We1pk,
                            const __bf16* __restrict__ We2pk,
                            float* __restrict__ ws_sum,
                            float* __restrict__ ws_cnt,
                            int E) {
  __shared__ WaveLds smem[4];
  const int lane = threadIdx.x & 31;
  const int wave = threadIdx.x >> 5;
  WaveLds& L = smem[wave];

  const int tile = blockIdx.x * 4 + wave;
  const int e0   = tile * 16;

  const int col   = lane & 15;    // WMMA B/C column, also edge-in-tile id
  const int hf    = lane >> 4;    // lane half
  const int rbase = hf * 8;       // C-layout row base for this lane

  // ---- per-tile geometric coefficients into LDS (each lane: 1 edge, 8 m) --
  {
    const int e = e0 + col;
    const bool v = (e < E);
    float s2 = 0.f, v2x = 0.f, v2y = 0.f, v2z = 0.f;
    if (v) {
      const float* sp = sh + (size_t)e * 4;
      s2 = sp[0]; v2x = sp[1]; v2y = sp[2]; v2z = sp[3];
    }
    const float* xp = x_src + (size_t)e * 64;
    #pragma unroll
    for (int mm = 0; mm < 8; ++mm) {
      const int m = rbase + mm;
      float s1 = 0.f, vx = 0.f, vy = 0.f, vz = 0.f;
      if (v) {
        s1 = xp[m];
        vx = xp[16 + m * 3 + 0];
        vy = xp[16 + m * 3 + 1];
        vz = xp[16 + m * 3 + 2];
      }
      const float dot = vx * v2x + vy * v2y + vz * v2z;
      const float cx = vy * v2z - vz * v2y;
      const float cy = vz * v2x - vx * v2z;
      const float cz = vx * v2y - vy * v2x;
      L.coef0[0][m][col] = C_A0 * s1 * s2;
      L.coef0[1][m][col] = (C_A0 * C_INV3) * dot;
      L.coef1[0][0][m][col] = C_A1 * s1 * v2x;
      L.coef1[0][1][m][col] = C_A1 * s1 * v2y;
      L.coef1[0][2][m][col] = C_A1 * s1 * v2z;
      L.coef1[1][0][m][col] = C_A1 * s2 * vx;
      L.coef1[1][1][m][col] = C_A1 * s2 * vy;
      L.coef1[1][2][m][col] = C_A1 * s2 * vz;
      L.coef1[2][0][m][col] = (C_A1 * C_INV2) * cx;
      L.coef1[2][1][m][col] = (C_A1 * C_INV2) * cy;
      L.coef1[2][2][m][col] = (C_A1 * C_INV2) * cz;
    }
  }

  // ---- GEMM1 A operand: edge_attr tile (16x32 bf16, ISA A layout) ---------
  // lane row = lane&15; K(h) = kb + h + (h>=8 ? 8 : 0), kb = (lane>>4)*8
  bf16x16 aop1;
  {
    const int e = e0 + col;
    const int kb = hf * 8;
    const float* p = edge_attr + (size_t)e * ED;
    #pragma unroll
    for (int h = 0; h < 16; ++h) {
      const int K = kb + h + (h >= 8 ? 8 : 0);
      const float f = (e < E) ? p[K] : 0.0f;
      aop1[h] = (__bf16)f;
    }
  }

  // ---- GEMM1: H = relu(EA @ We1 + be1) -> LDS (bf16) ----------------------
  #pragma unroll
  for (int n1 = 0; n1 < 8; ++n1) {
    bf16x16 b = *(const bf16x16*)(We1pk + ((n1 * 32 + lane) << 4));
    const float bias = be1[n1 * 16 + col];
    f32x8 c;
    #pragma unroll
    for (int r = 0; r < 8; ++r) c[r] = bias;
    c = __builtin_amdgcn_wmma_f32_16x16x32_bf16(false, aop1, false, b,
                                                (short)0, c, false, false);
    #pragma unroll
    for (int r = 0; r < 8; ++r) {
      const float hv = fmaxf(c[r], 0.0f);
      L.H[rbase + r][n1 * 16 + col] = (__bf16)hv;
    }
  }

  // ---- GEMM2 A operands: 4 K-slices of H (reused across all 112 chunks) ---
  bf16x16 a2[4];
  #pragma unroll
  for (int kk = 0; kk < 4; ++kk) {
    const __bf16* hp = &L.H[col][kk * 32 + hf * 8];
    const bf16x8 lo = *(const bf16x8*)hp;         // K = kk*32+kb   .. +7
    const bf16x8 hi = *(const bf16x8*)(hp + 16);  // K = kk*32+kb+16.. +7
    #pragma unroll
    for (int h = 0; h < 8; ++h) { a2[kk][h] = lo[h]; a2[kk][8 + h] = hi[h]; }
  }

  // ---- main loop: 112 chunks of w, 4 WMMAs each, fused contraction --------
  f32x8 o00, o01, o1x, o1y, o1z;
  #pragma unroll
  for (int r = 0; r < 8; ++r) { o00[r]=0.f; o01[r]=0.f; o1x[r]=0.f; o1y[r]=0.f; o1z[r]=0.f; }

  // chunk evaluator: one 16-wide slice of w for 16 edges (4 K-steps of WMMA)
  auto wchunk = [&](int n) -> f32x8 {
    const float bias = be2[n * 16 + col];
    f32x8 acc;
    #pragma unroll
    for (int r = 0; r < 8; ++r) acc[r] = bias;
    const __bf16* bp = We2pk + (((size_t)n * 4) << 9) + (lane << 4);
    acc = __builtin_amdgcn_wmma_f32_16x16x32_bf16(false, a2[0], false,
            *(const bf16x16*)(bp), (short)0, acc, false, false);
    acc = __builtin_amdgcn_wmma_f32_16x16x32_bf16(false, a2[1], false,
            *(const bf16x16*)(bp + 512), (short)0, acc, false, false);
    acc = __builtin_amdgcn_wmma_f32_16x16x32_bf16(false, a2[2], false,
            *(const bf16x16*)(bp + 1024), (short)0, acc, false, false);
    acc = __builtin_amdgcn_wmma_f32_16x16x32_bf16(false, a2[3], false,
            *(const bf16x16*)(bp + 1536), (short)0, acc, false, false);
    return acc;
  };
  auto axpy = [&](f32x8& d, const float* cp, f32x8 w) {
    const f32x4 cA = *(const f32x4*)cp;
    const f32x4 cB = *(const f32x4*)(cp + 4);
    #pragma unroll
    for (int r = 0; r < 4; ++r) d[r]     = fmaf(cA[r], w[r],     d[r]);
    #pragma unroll
    for (int r = 0; r < 4; ++r) d[4 + r] = fmaf(cB[r], w[4 + r], d[4 + r]);
  };

  // w_ss: chunks 0..31 -> out0 (tile n&1), coef0[0][m]
  for (int m = 0; m < 16; ++m) {
    axpy(o00, &L.coef0[0][m][rbase], wchunk(2 * m));
    axpy(o01, &L.coef0[0][m][rbase], wchunk(2 * m + 1));
  }
  // w_vv0: chunks 32..63 -> out0, coef0[1][m]
  for (int m = 0; m < 16; ++m) {
    axpy(o00, &L.coef0[1][m][rbase], wchunk(32 + 2 * m));
    axpy(o01, &L.coef0[1][m][rbase], wchunk(32 + 2 * m + 1));
  }
  // w_sv / w_vs / w_vv1: chunks 64..111 -> out1 (3 c-components)
  for (int blk = 0; blk < 3; ++blk) {
    for (int m = 0; m < 16; ++m) {
      const f32x8 w = wchunk(64 + blk * 16 + m);
      axpy(o1x, &L.coef1[blk][0][m][rbase], w);
      axpy(o1y, &L.coef1[blk][1][m][rbase], w);
      axpy(o1z, &L.coef1[blk][2][m][rbase], w);
    }
  }

  // ---- scatter messages: msg[e] = [out0(32) | out1(o,c)(48)] --------------
  #pragma unroll
  for (int r = 0; r < 8; ++r) {
    const int e = e0 + rbase + r;
    if (e < E) {
      const int d = dst[e];
      float* bp2 = ws_sum + (size_t)d * 80;
      atomicAdd(bp2 + col,              o00[r]);
      atomicAdd(bp2 + 16 + col,         o01[r]);
      atomicAdd(bp2 + 32 + col * 3 + 0, o1x[r]);
      atomicAdd(bp2 + 32 + col * 3 + 1, o1y[r]);
      atomicAdd(bp2 + 32 + col * 3 + 2, o1z[r]);
    }
  }
  if (lane < 16) {
    const int e = e0 + lane;
    if (e < E) atomicAdd(ws_cnt + dst[e], 1.0f);
  }
}

// ------------------------------ node finalize ------------------------------
__global__ void finalize_kernel(const float* __restrict__ ws_sum,
                                const float* __restrict__ ws_cnt,
                                const float* __restrict__ x_dst,
                                const float* __restrict__ Wr0,
                                const float* __restrict__ Wr1,
                                float* __restrict__ out, int N) {
  const int n = blockIdx.x * blockDim.x + threadIdx.x;
  if (n >= N) return;
  const float inv = 1.0f / fmaxf(ws_cnt[n], 1.0f);
  const float* ms = ws_sum + (size_t)n * 80;
  const float* xd = x_dst + (size_t)n * 64;

  float msg0[32];
  #pragma unroll
  for (int j = 0; j < 32; ++j) msg0[j] = ms[j] * inv;
  for (int o = 0; o < 32; ++o) {
    float a = 0.f;
    #pragma unroll
    for (int m = 0; m < 16; ++m) a += xd[m] * Wr0[m * 32 + o];
    msg0[o] += 0.25f * a;             // 1/sqrt(M) = 1/4
  }
  float* op = out + (size_t)n * 64;
  #pragma unroll
  for (int j = 0; j < 16; ++j) op[j] = fmaxf(msg0[j], 0.0f);
  for (int o = 0; o < 16; ++o) {
    const float g = 1.0f / (1.0f + expf(-msg0[16 + o]));
    #pragma unroll
    for (int c = 0; c < 3; ++c) {
      float mv = ms[32 + o * 3 + c] * inv;
      float a = 0.f;
      #pragma unroll
      for (int m = 0; m < 16; ++m) a += xd[16 + m * 3 + c] * Wr1[m * 16 + o];
      mv += 0.25f * a;
      op[16 + o * 3 + c] = mv * g;
    }
  }
}

// --------------------------------- launch ----------------------------------
extern "C" void kernel_launch(void* const* d_in, const int* in_sizes, int n_in,
                              void* d_out, int out_size, void* d_ws, size_t ws_size,
                              hipStream_t stream) {
  const int*   dst       = (const int*)  d_in[0];
  const float* x_src     = (const float*)d_in[1];
  const float* x_dst     = (const float*)d_in[2];
  const float* sh        = (const float*)d_in[3];
  const float* edge_attr = (const float*)d_in[4];
  const float* We1       = (const float*)d_in[5];
  const float* be1       = (const float*)d_in[6];
  const float* We2       = (const float*)d_in[7];
  const float* be2       = (const float*)d_in[8];
  const float* Wr0       = (const float*)d_in[9];
  const float* Wr1       = (const float*)d_in[10];
  (void)n_in; (void)out_size; (void)ws_size;

  const int E = in_sizes[0];
  const int N = in_sizes[2] / 64;

  // workspace carve-up (256B aligned)
  char* ws = (char*)d_ws;
  size_t off = 0;
  auto carve = [&](size_t bytes) {
    size_t o = off;
    off = (off + bytes + 255) & ~(size_t)255;
    return o;
  };
  float*  ws_sum = (float*) (ws + carve((size_t)N * 80 * sizeof(float)));
  float*  ws_cnt = (float*) (ws + carve((size_t)N * sizeof(float)));
  __bf16* We1pk  = (__bf16*)(ws + carve((size_t)8 * 32 * 16 * sizeof(__bf16)));
  __bf16* We2pk  = (__bf16*)(ws + carve((size_t)NCHUNK * 4 * 32 * 16 * sizeof(__bf16)));

  {
    const long long tot = (long long)N * 80;
    zero_f32<<<(unsigned)((tot + 255) / 256), 256, 0, stream>>>(ws_sum, tot);
    zero_f32<<<(unsigned)((N + 255) / 256), 256, 0, stream>>>(ws_cnt, (long long)N);
  }
  {
    const int tot = 8 * 32 * 16 + NCHUNK * 4 * 32 * 16;
    pack_weights<<<(tot + 255) / 256, 256, 0, stream>>>(We1, We2, We1pk, We2pk);
  }
  {
    const int tiles = (E + 15) / 16;
    const int blocks = (tiles + 3) / 4;   // 4 waves / block, 1 tile / wave
    edge_kernel<<<blocks, 128, 0, stream>>>(dst, x_src, sh, edge_attr, be1, be2,
                                            We1pk, We2pk, ws_sum, ws_cnt, E);
  }
  finalize_kernel<<<(N + 255) / 256, 256, 0, stream>>>(ws_sum, ws_cnt, x_dst,
                                                       Wr0, Wr1, (float*)d_out, N);
}
